// CosmicBaseModel_2181843387116
// MI455X (gfx1250) — matrix operand order repaired and checked by reading, not compiled, not run on hardware
//
#include <hip/hip_runtime.h>
#include <hip/hip_bf16.h>

typedef __attribute__((ext_vector_type(16))) _Float16 v16h;
typedef __attribute__((ext_vector_type(8)))  _Float16 v8h;
typedef __attribute__((ext_vector_type(8)))  float    v8f;

#define HIDDEN 256
#define COSMIC 512
#define BATCH  8
#define SEQ    2048
#define MTOT   (BATCH * SEQ)   // 16384

union AFrag { v16h v; v8h h[2]; };

// A fragment (16x32 f16, M x K): per lane: row = lane&15,
// K chunks [khalf..khalf+7] and [khalf+16..khalf+23], khalf = (lane<16)?0:8.
__device__ __forceinline__ v16h load_a_f16(const _Float16* base, int row_base, int k_base, int ld) {
  const int lane  = threadIdx.x & 31;
  const int m     = lane & 15;
  const int khalf = (lane & 16) ? 8 : 0;
  const _Float16* p = base + (size_t)(row_base + m) * ld + (k_base + khalf);
  AFrag f;
  f.h[0] = *(const v8h*)p;
  f.h[1] = *(const v8h*)(p + 16);
  return f.v;
}

__device__ __forceinline__ v8h cvt8_f32(const float* p) {
  v8h r;
#pragma unroll
  for (int i = 0; i < 8; ++i) r[i] = (_Float16)p[i];
  return r;
}

__device__ __forceinline__ v16h load_a_f32(const float* base, int row_base, int k_base, int ld) {
  const int lane  = threadIdx.x & 31;
  const int m     = lane & 15;
  const int khalf = (lane & 16) ? 8 : 0;
  const float* p = base + (size_t)(row_base + m) * ld + (k_base + khalf);
  AFrag f;
  f.h[0] = cvt8_f32(p);
  f.h[1] = cvt8_f32(p + 16);
  return f.v;
}

// B fragment (32x16 f16, K x N) from BT = row-major [N][K] (column-major B):
// per lane: column n = lane&15, K chunks [kofs..kofs+7] and [kofs+16..kofs+23], kofs = (lane<16)?0:8.
__device__ __forceinline__ v16h load_b_f16(const _Float16* bt, int n_base, int k_base, int ld) {
  const int lane = threadIdx.x & 31;
  const int n    = lane & 15;
  const int kofs = (lane & 16) ? 8 : 0;
  const _Float16* p = bt + (size_t)(n_base + n) * ld + (k_base + kofs);
  AFrag f;
  f.h[0] = *(const v8h*)p;
  f.h[1] = *(const v8h*)(p + 16);
  return f.v;
}

__device__ __forceinline__ v8f wmma_f16(v16h a, v16h b, v8f c) {
  return __builtin_amdgcn_wmma_f32_16x16x32_f16(false, a, false, b, (short)0, c, false, false);
}

// ---------------------------------------------------------------------------
// Weight prep: f32 -> f16 with transpose into BT layout.
// WencT[n][k] = W_enc[k][n]  (512 x 256),  WdecT[n][k] = W_dec[k][n] (256 x 512)
// ---------------------------------------------------------------------------
__global__ void prep_weights(const float* __restrict__ Wenc, const float* __restrict__ Wdec,
                             _Float16* __restrict__ WencT, _Float16* __restrict__ WdecT) {
  const int idx = blockIdx.x * blockDim.x + threadIdx.x;
  if (idx < HIDDEN * COSMIC) {
    const int n  = idx / HIDDEN, k  = idx % HIDDEN;
    WencT[idx] = (_Float16)Wenc[(size_t)k * COSMIC + n];
    const int n2 = idx / COSMIC, k2 = idx % COSMIC;
    WdecT[idx] = (_Float16)Wdec[(size_t)k2 * HIDDEN + n2];
  }
}

// ---------------------------------------------------------------------------
// Encode: E = X @ W_enc + b_enc ; writes E16 [M][512] and ET16 [B][512][SEQ]
// 16x64 output tile per wave: A-fragment reused across 4 WMMAs per k-step.
// ---------------------------------------------------------------------------
__global__ void encode_kernel(const float* __restrict__ X, const _Float16* __restrict__ WencT,
                              const float* __restrict__ b_enc,
                              _Float16* __restrict__ E16, _Float16* __restrict__ ET16) {
  const int wave   = blockIdx.x * (blockDim.x >> 5) + (threadIdx.x >> 5);
  const int nsuper = COSMIC / 64;                 // 8
  const int m0 = (wave / nsuper) * 16;
  const int n0 = (wave % nsuper) * 64;

  v8f acc[4] = {{}, {}, {}, {}};
#pragma unroll
  for (int k = 0; k < HIDDEN; k += 32) {
    v16h a = load_a_f32(X, m0, k, HIDDEN);
#pragma unroll
    for (int i = 0; i < 4; ++i) {
      v16h b = load_b_f16(WencT, n0 + 16 * i, k, HIDDEN);
      acc[i] = wmma_f16(a, b, acc[i]);
    }
  }

  const int lane  = threadIdx.x & 31;
  const int mbase = (lane & 16) ? 8 : 0;
  const int bb = m0 >> 11;                        // batch index (tiles never straddle batches)
  const int s0 = (m0 & (SEQ - 1)) + mbase;

#pragma unroll
  for (int i = 0; i < 4; ++i) {
    const int ncol = n0 + 16 * i + (lane & 15);
    const float bias = b_enc[ncol];
    v8h hv;
#pragma unroll
    for (int v = 0; v < 8; ++v) {
      const _Float16 h = (_Float16)(acc[i][v] + bias);
      hv[v] = h;
      E16[(size_t)(m0 + mbase + v) * COSMIC + ncol] = h;
    }
    // lane holds one column of C over 8 consecutive rows -> contiguous v8h store
    *(v8h*)(ET16 + ((size_t)bb * COSMIC + ncol) * SEQ + s0) = hv;
  }
}

// ---------------------------------------------------------------------------
// Attention: per (batch, 16-query block). 8 waves, 256 threads.
//   Pass 1: scores[16][2048] = Q @ K^T into LDS (f32); Q frags held in registers
//   Pass 2: softmax (f32, shfl width-16 reductions); writes normalized f16 probs
//   Pass 3: attended = P(f16, LDS) @ E (B-frags from ET16), store f16 to A16
// Dynamic LDS: 128 KB scores + 64 KB probs = 192 KB (within 320 KB WGP budget).
// ---------------------------------------------------------------------------
__global__ void attn_kernel(const _Float16* __restrict__ E16, const _Float16* __restrict__ ET16,
                            _Float16* __restrict__ A16) {
  extern __shared__ char smem[];
  float*    sc  = (float*)smem;                        // [16][SEQ] f32 scores
  _Float16* p16 = (_Float16*)(smem + 16 * SEQ * 4);    // [16][SEQ] f16 probs

  const int bb = blockIdx.x >> 7;                 // / (SEQ/16)
  const int q0 = (blockIdx.x & 127) * 16;
  const _Float16* Eb  = E16  + (size_t)bb * SEQ * COSMIC;
  const _Float16* ETb = ET16 + (size_t)bb * COSMIC * SEQ;
  const int wave = threadIdx.x >> 5;
  const int lane = threadIdx.x & 31;

  // ---- Pass 1: hoist Q fragments (16x512 block) into registers, then stream keys ----
  v16h q[16];
#pragma unroll
  for (int kk = 0; kk < 16; ++kk)
    q[kk] = load_a_f16(Eb, q0, kk * 32, COSMIC);

  for (int kt = 0; kt < 16; ++kt) {
    const int t0 = (wave * 16 + kt) * 16;
    v8f acc = {};
#pragma unroll
    for (int kk = 0; kk < 16; ++kk) {
      v16h b = load_b_f16(Eb, t0, kk * 32, COSMIC);   // keys: B column n = token t0+n = E row
      acc = wmma_f16(q[kk], b, acc);
    }
    const int n     = lane & 15;
    const int mbase = (lane & 16) ? 8 : 0;
#pragma unroll
    for (int v = 0; v < 8; ++v)
      sc[(mbase + v) * SEQ + t0 + n] = acc[v];
  }
  __syncthreads();

  // ---- Pass 2: softmax, 16 threads per row (contiguous lanes -> shfl width 16) ----
  {
    const int r   = threadIdx.x >> 4;             // 0..15
    const int seg = threadIdx.x & 15;             // 0..15
    float* row = sc + r * SEQ + seg * 128;
    float mx = -3.4e38f;
#pragma unroll 4
    for (int j = 0; j < 128; ++j) mx = fmaxf(mx, row[j]);
#pragma unroll
    for (int off = 8; off; off >>= 1) mx = fmaxf(mx, __shfl_xor(mx, off, 16));
    float sum = 0.0f;
#pragma unroll 4
    for (int j = 0; j < 128; ++j) { const float p = __expf(row[j] - mx); row[j] = p; sum += p; }
#pragma unroll
    for (int off = 8; off; off >>= 1) sum += __shfl_xor(sum, off, 16);
    const float inv = 1.0f / sum;
    _Float16* prow = p16 + r * SEQ + seg * 128;
#pragma unroll 4
    for (int j = 0; j < 128; ++j) prow[j] = (_Float16)(row[j] * inv);
  }
  __syncthreads();

  // ---- Pass 3: attended = P @ E ; each wave owns 64 output dims ----
  const int n0 = wave * 64;
  v8f acc3[4] = {{}, {}, {}, {}};
  const int m     = lane & 15;
  const int khalf = (lane & 16) ? 8 : 0;
  for (int t = 0; t < SEQ; t += 32) {
    // A fragment: f16 probabilities straight from LDS (two 16B ds loads, no cvt)
    const _Float16* pp = p16 + m * SEQ + t + khalf;
    AFrag f;
    f.h[0] = *(const v8h*)pp;
    f.h[1] = *(const v8h*)(pp + 16);
#pragma unroll
    for (int i = 0; i < 4; ++i) {
      v16h b = load_b_f16(ETb, n0 + 16 * i, t, SEQ);  // B column n = dim, K = token (contig in ET)
      acc3[i] = wmma_f16(f.v, b, acc3[i]);
    }
  }
  const int nn    = lane & 15;
  const int mbase = (lane & 16) ? 8 : 0;
#pragma unroll
  for (int i = 0; i < 4; ++i) {
    const int ncol = n0 + 16 * i + nn;
#pragma unroll
    for (int v = 0; v < 8; ++v)
      A16[(size_t)(bb * SEQ + q0 + mbase + v) * COSMIC + ncol] = (_Float16)acc3[i][v];
  }
}

// ---------------------------------------------------------------------------
// Decode: out = A @ W_dec + b_dec (f32 output). 16x64 tile per wave.
// ---------------------------------------------------------------------------
__global__ void decode_kernel(const _Float16* __restrict__ A16, const _Float16* __restrict__ WdecT,
                              const float* __restrict__ b_dec, float* __restrict__ out) {
  const int wave   = blockIdx.x * (blockDim.x >> 5) + (threadIdx.x >> 5);
  const int nsuper = HIDDEN / 64;                 // 4
  const int m0 = (wave / nsuper) * 16;
  const int n0 = (wave % nsuper) * 64;

  v8f acc[4] = {{}, {}, {}, {}};
#pragma unroll
  for (int k = 0; k < COSMIC; k += 32) {
    v16h a = load_a_f16(A16, m0, k, COSMIC);
#pragma unroll
    for (int i = 0; i < 4; ++i) {
      v16h b = load_b_f16(WdecT, n0 + 16 * i, k, COSMIC);
      acc[i] = wmma_f16(a, b, acc[i]);
    }
  }

  const int lane  = threadIdx.x & 31;
  const int mbase = (lane & 16) ? 8 : 0;
#pragma unroll
  for (int i = 0; i < 4; ++i) {
    const int ncol = n0 + 16 * i + (lane & 15);
    const float bias = b_dec[ncol];
#pragma unroll
    for (int v = 0; v < 8; ++v)
      out[(size_t)(m0 + mbase + v) * HIDDEN + ncol] = acc[i][v] + bias;
  }
}

// ---------------------------------------------------------------------------
extern "C" void kernel_launch(void* const* d_in, const int* in_sizes, int n_in,
                              void* d_out, int out_size, void* d_ws, size_t ws_size,
                              hipStream_t stream) {
  (void)in_sizes; (void)n_in; (void)out_size; (void)ws_size;
  const float* X    = (const float*)d_in[0];     // [B,S,HIDDEN]
  const float* Wenc = (const float*)d_in[1];     // [HIDDEN,COSMIC]
  const float* benc = (const float*)d_in[2];     // [COSMIC]
  const float* Wdec = (const float*)d_in[3];     // [COSMIC,HIDDEN]
  const float* bdec = (const float*)d_in[4];     // [HIDDEN]
  float* out = (float*)d_out;                    // [B,S,HIDDEN] f32

  // Workspace carve-out (~48.5 MiB total)
  char* ws = (char*)d_ws;
  _Float16* E16   = (_Float16*)ws; ws += (size_t)MTOT * COSMIC * sizeof(_Float16);
  _Float16* ET16  = (_Float16*)ws; ws += (size_t)MTOT * COSMIC * sizeof(_Float16);
  _Float16* A16   = (_Float16*)ws; ws += (size_t)MTOT * COSMIC * sizeof(_Float16);
  _Float16* WencT = (_Float16*)ws; ws += (size_t)HIDDEN * COSMIC * sizeof(_Float16);
  _Float16* WdecT = (_Float16*)ws; ws += (size_t)HIDDEN * COSMIC * sizeof(_Float16);

  prep_weights<<<(HIDDEN * COSMIC + 255) / 256, 256, 0, stream>>>(Wenc, Wdec, WencT, WdecT);

  // 16384/16 M-tiles * 512/64 N-supertiles = 8192 wave-tiles, 8 waves/block
  encode_kernel<<<(MTOT / 16) * (COSMIC / 64) / 8, 256, 0, stream>>>(X, WencT, benc, E16, ET16);

  // one workgroup per (batch, 16-query block); 192 KB dynamic LDS
  attn_kernel<<<BATCH * (SEQ / 16), 256, 16 * SEQ * (sizeof(float) + sizeof(_Float16)), stream>>>(
      E16, ET16, A16);

  // 16384/16 M-tiles * 256/64 N-supertiles = 4096 wave-tiles, 8 waves/block
  decode_kernel<<<(MTOT / 16) * (HIDDEN / 64) / 8, 256, 0, stream>>>(A16, WdecT, bdec, out);
}
